// KimiDecoderLayer_3246995276384
// MI455X (gfx1250) — compile-verified, matrix operand order checked by reading, not run on hardware
//
#include <hip/hip_runtime.h>
#include <hip/hip_bf16.h>
#include <math.h>

// ---------------- model constants ----------------
#define TT   1024
#define DD   2048
#define HH   16
#define DKK  128
#define DVV  128
#define LOWW 128
#define II   8192

typedef __attribute__((ext_vector_type(16))) __bf16 v16bf;
typedef __attribute__((ext_vector_type(8)))  float  v8f;

struct alignas(16) U4 { unsigned int x, y, z, w; };
union FragCvt { U4 u[2]; v16bf v; };

__device__ __forceinline__ float sigmoidf_(float x) { return 1.0f / (1.0f + __expf(-x)); }
__device__ __forceinline__ float siluf_(float x)    { return x * sigmoidf_(x); }

// Load a 16x32 bf16 fragment for lane l from a row-major matrix (ISA 7.12.2 layout):
// lane's 16 values are two contiguous 8-element (16B) chunks at K = kk+half*8 and K = kk+16+half*8.
// Used for the A matrix (lane = output row) and for B^T rows (lane = output column).
__device__ __forceinline__ v16bf load_frag(const __bf16* __restrict__ row, int kk, int half) {
  const __bf16* p = row + kk + (half << 3);
  FragCvt t;
  t.u[0] = *(const U4*)(p);
  t.u[1] = *(const U4*)(p + 16);
  return t.v;
}

// ---------------- transpose + convert: B[K,N] f32 -> Bt[N,K] bf16 (LDS tiled) ----------------
__global__ __launch_bounds__(256) void transpose_bf16_kernel(const float* __restrict__ in,
                                                             __bf16* __restrict__ out,
                                                             int K, int N) {
  __shared__ float tile[32][33];
  int bk = blockIdx.y << 5, bn = blockIdx.x << 5;
  int tx = threadIdx.x, ty = threadIdx.y;           // 32 x 8
  for (int i = ty; i < 32; i += 8) {
    int k = bk + i, n = bn + tx;
    if (k < K && n < N) tile[i][tx] = in[(size_t)k * N + n];
  }
  __syncthreads();
  for (int i = ty; i < 32; i += 8) {
    int n = bn + i, k = bk + tx;
    if (n < N && k < K) out[(size_t)n * K + k] = (__bf16)tile[tx][i];
  }
}

// ---------------- RMSNorm (row per block), bf16 output for WMMA A operand ----------------
__global__ __launch_bounds__(256) void rmsnorm_kernel(const float* __restrict__ x,
                                                      const float* __restrict__ w,
                                                      __bf16* __restrict__ out) {
  int t = blockIdx.x, tid = threadIdx.x;
  __shared__ float red[256];
  const float* xr = x + (size_t)t * DD;
  float s = 0.f;
  for (int i = tid; i < DD; i += 256) { float v = xr[i]; s += v * v; }
  red[tid] = s; __syncthreads();
  for (int st = 128; st > 0; st >>= 1) { if (tid < st) red[tid] += red[tid + st]; __syncthreads(); }
  float inv = rsqrtf(red[0] / (float)DD + 1e-6f);
  __bf16* orow = out + (size_t)t * DD;
  for (int i = tid; i < DD; i += 256) orow[i] = (__bf16)(xr[i] * inv * w[i]);
}

// ---------------- WMMA GEMM, 16x64 tile per wave (4 accumulators share one A fragment) ----
// A: bf16 row-major [M,K]; Bt: bf16 row-major [N,K] (= B transposed); C: f32 [M,N].
// N % 64 == 0, K % 32 == 0. Inner loop: 10 x global_load_b128 + 4 x v_wmma.
__global__ __launch_bounds__(32) void gemm_wmma4(const __bf16* __restrict__ A,
                                                 const __bf16* __restrict__ Bt,
                                                 float* __restrict__ C,
                                                 int M, int N, int K) {
  int tn = blockIdx.x << 6, tm = blockIdx.y << 4;
  int l = threadIdx.x, half = l >> 4, lr = l & 15;
  const __bf16* Arow = A + (size_t)(tm + lr) * K;
  const __bf16* Btr[4];
#pragma unroll
  for (int f = 0; f < 4; ++f) Btr[f] = Bt + (size_t)(tn + (f << 4) + lr) * K;
  v8f acc[4] = {{}, {}, {}, {}};
  for (int kk = 0; kk < K; kk += 32) {
    if (kk + 32 < K) {
      __builtin_prefetch(Arow + kk + 32, 0, 1);
#pragma unroll
      for (int f = 0; f < 4; ++f) __builtin_prefetch(Btr[f] + kk + 32, 0, 1);
    }
    v16bf a = load_frag(Arow, kk, half);
#pragma unroll
    for (int f = 0; f < 4; ++f) {
      v16bf b = load_frag(Btr[f], kk, half);
      acc[f] = __builtin_amdgcn_wmma_f32_16x16x32_bf16(false, a, false, b, (short)0, acc[f], false, false);
    }
  }
#pragma unroll
  for (int f = 0; f < 4; ++f)
#pragma unroll
    for (int i = 0; i < 8; ++i)
      C[(size_t)(tm + i + (half << 3)) * N + (tn + (f << 4) + lr)] = acc[f][i];
}

// ---------------- WMMA GEMM, single 16x16 tile (for small N, e.g. N=16) ----------------
__global__ __launch_bounds__(32) void gemm_wmma1(const __bf16* __restrict__ A,
                                                 const __bf16* __restrict__ Bt,
                                                 float* __restrict__ C,
                                                 int M, int N, int K) {
  int tn = blockIdx.x << 4, tm = blockIdx.y << 4;
  int l = threadIdx.x, half = l >> 4, lr = l & 15;
  const __bf16* Arow = A + (size_t)(tm + lr) * K;
  const __bf16* Btrow = Bt + (size_t)(tn + lr) * K;
  v8f acc = {};
  for (int kk = 0; kk < K; kk += 32) {
    v16bf a = load_frag(Arow, kk, half);
    v16bf b = load_frag(Btrow, kk, half);
    acc = __builtin_amdgcn_wmma_f32_16x16x32_bf16(false, a, false, b, (short)0, acc, false, false);
  }
#pragma unroll
  for (int i = 0; i < 8; ++i)
    C[(size_t)(tm + i + (half << 3)) * N + (tn + lr)] = acc[i];
}

// ---------------- fused SwiGLU dual-GEMM: P = silu(A@Bg) * (A@Bu), bf16 out ----------------
// 16x32 tile per wave: 2 N-tiles x {gate,up} = 4 WMMAs sharing one A fragment. N % 32 == 0.
__global__ __launch_bounds__(32) void gemm_swiglu_wmma(const __bf16* __restrict__ A,
                                                       const __bf16* __restrict__ Bgt,
                                                       const __bf16* __restrict__ But,
                                                       __bf16* __restrict__ P,
                                                       int M, int N, int K) {
  int tn = blockIdx.x << 5, tm = blockIdx.y << 4;
  int l = threadIdx.x, half = l >> 4, lr = l & 15;
  const __bf16* Arow = A + (size_t)(tm + lr) * K;
  const __bf16* Bgr[2] = { Bgt + (size_t)(tn + lr) * K, Bgt + (size_t)(tn + 16 + lr) * K };
  const __bf16* Bur[2] = { But + (size_t)(tn + lr) * K, But + (size_t)(tn + 16 + lr) * K };
  v8f accg[2] = {{}, {}}, accu[2] = {{}, {}};
  for (int kk = 0; kk < K; kk += 32) {
    if (kk + 32 < K) {
      __builtin_prefetch(Arow + kk + 32, 0, 1);
      __builtin_prefetch(Bgr[0] + kk + 32, 0, 1);
      __builtin_prefetch(Bur[0] + kk + 32, 0, 1);
    }
    v16bf a = load_frag(Arow, kk, half);
#pragma unroll
    for (int f = 0; f < 2; ++f) {
      v16bf bg = load_frag(Bgr[f], kk, half);
      accg[f] = __builtin_amdgcn_wmma_f32_16x16x32_bf16(false, a, false, bg, (short)0, accg[f], false, false);
      v16bf bu = load_frag(Bur[f], kk, half);
      accu[f] = __builtin_amdgcn_wmma_f32_16x16x32_bf16(false, a, false, bu, (short)0, accu[f], false, false);
    }
  }
#pragma unroll
  for (int f = 0; f < 2; ++f)
#pragma unroll
    for (int i = 0; i < 8; ++i) {
      float g = accg[f][i], u = accu[f][i];
      P[(size_t)(tm + i + (half << 3)) * N + (tn + (f << 4) + lr)] = (__bf16)(siluf_(g) * u);
    }
}

// ---------------- f32 -> bf16 convert (for GEMM-produced A operands) ----------------
__global__ void cvt_bf16_kernel(const float* __restrict__ in, __bf16* __restrict__ out, int n) {
  int i = blockIdx.x * blockDim.x + threadIdx.x;
  if (i < n) out[i] = (__bf16)in[i];
}

// ---------------- causal depthwise conv (K=4) + SiLU ----------------
__global__ void conv_silu_kernel(const float* __restrict__ in,
                                 const float* __restrict__ w,
                                 float* __restrict__ out, int C) {
  int idx = blockIdx.x * blockDim.x + threadIdx.x;
  if (idx >= TT * C) return;
  int c = idx % C, t = idx / C;
  float s = 0.f;
#pragma unroll
  for (int i = 0; i < 4; ++i) {
    int tt = t - 3 + i;
    if (tt >= 0) s += w[i * C + c] * in[(size_t)tt * C + c];
  }
  out[idx] = siluf_(s);
}

// ---------------- per-(t,h) L2 norm over 128, optional extra scale ----------------
__global__ __launch_bounds__(128) void l2norm_kernel(float* __restrict__ x, float scale) {
  int r = blockIdx.x, tid = threadIdx.x;
  __shared__ float red[128];
  float v = x[(size_t)r * 128 + tid];
  red[tid] = v * v; __syncthreads();
  for (int st = 64; st > 0; st >>= 1) { if (tid < st) red[tid] += red[tid + st]; __syncthreads(); }
  x[(size_t)r * 128 + tid] = v * rsqrtf(red[0] + 1e-6f) * scale;
}

// ---------------- g = -exp(a_log)*softplus(fa@wfb + dt_bias); beta = sigmoid(braw) ----------------
__global__ void gbeta_kernel(const float* __restrict__ fa, const float* __restrict__ wfb,
                             const float* __restrict__ dt_bias, const float* __restrict__ a_log,
                             const float* __restrict__ braw,
                             float* __restrict__ g, float* __restrict__ beta) {
  int idx = blockIdx.x * blockDim.x + threadIdx.x;
  if (idx >= TT * HH) return;
  int hh = idx & (HH - 1), t = idx >> 4;
  float s = dt_bias[hh];
  const float* far = fa + (size_t)t * LOWW;
  for (int j = 0; j < LOWW; ++j) s += far[j] * wfb[j * HH + hh];
  float sp = (s > 20.f) ? s : log1pf(__expf(s));
  g[idx] = -__expf(a_log[hh]) * sp;
  beta[idx] = sigmoidf_(braw[idx]);
}

// ---------------- gated delta-rule scan: one head per workgroup, S resident in LDS ----------------
__global__ __launch_bounds__(256) void kda_scan_kernel(const float* __restrict__ q,
                                                       const float* __restrict__ k,
                                                       const float* __restrict__ v,
                                                       const float* __restrict__ g,
                                                       const float* __restrict__ beta,
                                                       float* __restrict__ o) {
  __shared__ float S[DKK * DVV];          // 64 KB state, fits in 320 KB WGP LDS
  __shared__ float ks[DKK], qs[DKK], vs[DVV], dvs[DVV];
  __shared__ float part[256];
  __shared__ float s_dec, s_beta;
  int h = blockIdx.x, tid = threadIdx.x;
  int vcol = tid & 127, kh = tid >> 7;    // each thread: fixed v-column, half of k-rows
  for (int i = tid; i < DKK * DVV; i += 256) S[i] = 0.f;
  __syncthreads();
  for (int t = 0; t < TT; ++t) {
    size_t base = ((size_t)t * HH + h) * 128;
    if (tid < 128) { ks[tid] = k[base + tid]; qs[tid] = q[base + tid]; vs[tid] = v[base + tid]; }
    if (tid == 0) { s_dec = __expf(g[t * HH + h]); s_beta = beta[t * HH + h]; }
    __syncthreads();
    // phase 1: decay state, partial kS
    float dec = s_dec, acc = 0.f;
    for (int j = 0; j < 64; ++j) {
      int kr = (kh << 6) + j;
      float s = S[kr * DVV + vcol] * dec;
      S[kr * DVV + vcol] = s;
      acc += ks[kr] * s;
    }
    part[tid] = acc; __syncthreads();
    if (kh == 0) {
      float kS = part[vcol] + part[128 + vcol];
      dvs[vcol] = s_beta * (vs[vcol] - kS);
    }
    __syncthreads();
    // phase 2: rank-1 update, partial qS
    float d = dvs[vcol], acc2 = 0.f;
    for (int j = 0; j < 64; ++j) {
      int kr = (kh << 6) + j;
      float s = S[kr * DVV + vcol] + ks[kr] * d;
      S[kr * DVV + vcol] = s;
      acc2 += qs[kr] * s;
    }
    part[tid] = acc2; __syncthreads();
    if (kh == 0) o[base + vcol] = part[vcol] + part[128 + vcol];
    __syncthreads();
  }
}

// ---------------- gated RMSNorm on o: out = rmsnorm(o)*w * silu(gate), bf16 out ----------------
__global__ __launch_bounds__(128) void gated_out_kernel(const float* __restrict__ o,
                                                        const float* __restrict__ w,
                                                        const float* __restrict__ gate,
                                                        __bf16* __restrict__ out) {
  int r = blockIdx.x, tid = threadIdx.x;
  __shared__ float red[128];
  size_t idx = (size_t)r * 128 + tid;
  float v = o[idx];
  red[tid] = v * v; __syncthreads();
  for (int st = 64; st > 0; st >>= 1) { if (tid < st) red[tid] += red[tid + st]; __syncthreads(); }
  float inv = rsqrtf(red[0] / 128.f + 1e-6f);
  float gt = gate[idx];
  out[idx] = (__bf16)(v * inv * w[tid] * siluf_(gt));
}

// ---------------- elementwise residual add ----------------
__global__ void add_kernel(const float* __restrict__ a, const float* __restrict__ b,
                           float* __restrict__ c, int n) {
  int i = blockIdx.x * blockDim.x + threadIdx.x;
  if (i < n) c[i] = a[i] + b[i];
}

extern "C" void kernel_launch(void* const* d_in, const int* in_sizes, int n_in,
                              void* d_out, int out_size, void* d_ws, size_t ws_size,
                              hipStream_t stream) {
  const float* x        = (const float*)d_in[0];
  const float* wq       = (const float*)d_in[1];
  const float* wk       = (const float*)d_in[2];
  const float* wv       = (const float*)d_in[3];
  const float* conv_q   = (const float*)d_in[4];
  const float* conv_k   = (const float*)d_in[5];
  const float* conv_v   = (const float*)d_in[6];
  const float* wfa      = (const float*)d_in[7];
  const float* wfb      = (const float*)d_in[8];
  const float* wb       = (const float*)d_in[9];
  const float* wga      = (const float*)d_in[10];
  const float* wgb      = (const float*)d_in[11];
  const float* a_log    = (const float*)d_in[12];
  const float* dt_bias  = (const float*)d_in[13];
  const float* o_norm_w = (const float*)d_in[14];
  const float* wo       = (const float*)d_in[15];
  const float* ln1_w    = (const float*)d_in[16];
  const float* ln2_w    = (const float*)d_in[17];
  const float* w_gate   = (const float*)d_in[18];
  const float* w_up     = (const float*)d_in[19];
  const float* w_down   = (const float*)d_in[20];
  float* out = (float*)d_out;

  // ---- workspace layout ----
  float* ws = (float*)d_ws;
  const size_t TD = (size_t)TT * DD;       // 2M elements
  size_t off = 0;
  float* qpre  = ws + off; off += TD;
  float* kpre  = ws + off; off += TD;
  float* vpre  = ws + off; off += TD;
  float* qc    = ws + off; off += TD;
  float* kc    = ws + off; off += TD;
  float* vc    = ws + off; off += TD;
  float* gate  = ws + off; off += TD;
  float* o_s   = ws + off; off += TD;
  float* fa    = ws + off; off += (size_t)TT * LOWW;
  float* gaf   = ws + off; off += (size_t)TT * LOWW;
  float* braw  = ws + off; off += (size_t)TT * HH;
  float* garr  = ws + off; off += (size_t)TT * HH;
  float* barr  = ws + off; off += (size_t)TT * HH;
  // f32 buffer reuse: qpre/kpre/vpre are dead after the convs
  float* attn = qpre;
  float* x2   = vpre;
  float* mlp  = kpre;
  // bf16 region
  __bf16* bb   = (__bf16*)(ws + off);
  size_t boff = 0;
  __bf16* h_b   = bb + boff; boff += TD;                  // rmsnorm1 out
  __bf16* h2_b  = bb + boff; boff += TD;                  // rmsnorm2 out
  __bf16* ogt_b = bb + boff; boff += TD;                  // gated o
  __bf16* ga_b  = bb + boff; boff += (size_t)TT * LOWW;   // gate low-rank
  __bf16* P_b   = bb + boff; boff += (size_t)TT * II;     // swiglu out
  // bf16 transposed weight staging
  __bf16* wqT   = bb + boff; boff += (size_t)DD * DD;
  __bf16* wkT   = bb + boff; boff += (size_t)DD * DD;
  __bf16* wvT   = bb + boff; boff += (size_t)DD * DD;
  __bf16* woT   = bb + boff; boff += (size_t)DD * DD;
  __bf16* wfaT  = bb + boff; boff += (size_t)DD * LOWW;
  __bf16* wbT   = bb + boff; boff += (size_t)DD * HH;
  __bf16* wgaT  = bb + boff; boff += (size_t)DD * LOWW;
  __bf16* wgbT  = bb + boff; boff += (size_t)LOWW * DD;
  __bf16* wgT   = bb + boff; boff += (size_t)DD * II;
  __bf16* wuT   = bb + boff; boff += (size_t)DD * II;
  __bf16* wdT   = bb + boff; boff += (size_t)II * DD;
  (void)ws_size; (void)in_sizes; (void)n_in; (void)out_size;

  dim3 wave(32);
  dim3 tb(32, 8);
  int nTD = TT * DD;
  auto grid_t = [](int K, int N) { return dim3((N + 31) / 32, (K + 31) / 32); };

  // ---- 0. stage weights: f32 [K,N] -> bf16 transposed [N,K] ----
  transpose_bf16_kernel<<<grid_t(DD, DD), tb, 0, stream>>>(wq, wqT, DD, DD);
  transpose_bf16_kernel<<<grid_t(DD, DD), tb, 0, stream>>>(wk, wkT, DD, DD);
  transpose_bf16_kernel<<<grid_t(DD, DD), tb, 0, stream>>>(wv, wvT, DD, DD);
  transpose_bf16_kernel<<<grid_t(DD, DD), tb, 0, stream>>>(wo, woT, DD, DD);
  transpose_bf16_kernel<<<grid_t(DD, LOWW), tb, 0, stream>>>(wfa, wfaT, DD, LOWW);
  transpose_bf16_kernel<<<grid_t(DD, HH), tb, 0, stream>>>(wb, wbT, DD, HH);
  transpose_bf16_kernel<<<grid_t(DD, LOWW), tb, 0, stream>>>(wga, wgaT, DD, LOWW);
  transpose_bf16_kernel<<<grid_t(LOWW, DD), tb, 0, stream>>>(wgb, wgbT, LOWW, DD);
  transpose_bf16_kernel<<<grid_t(DD, II), tb, 0, stream>>>(w_gate, wgT, DD, II);
  transpose_bf16_kernel<<<grid_t(DD, II), tb, 0, stream>>>(w_up, wuT, DD, II);
  transpose_bf16_kernel<<<grid_t(II, DD), tb, 0, stream>>>(w_down, wdT, II, DD);

  // ---- 1. pre-attention RMSNorm (bf16 A operand) ----
  rmsnorm_kernel<<<TT, 256, 0, stream>>>(x, ln1_w, h_b);

  // ---- 2. WMMA projections off h ----
  gemm_wmma4<<<dim3(DD / 64, TT / 16), wave, 0, stream>>>(h_b, wqT, qpre, TT, DD, DD);
  gemm_wmma4<<<dim3(DD / 64, TT / 16), wave, 0, stream>>>(h_b, wkT, kpre, TT, DD, DD);
  gemm_wmma4<<<dim3(DD / 64, TT / 16), wave, 0, stream>>>(h_b, wvT, vpre, TT, DD, DD);
  gemm_wmma4<<<dim3(LOWW / 64, TT / 16), wave, 0, stream>>>(h_b, wfaT, fa, TT, LOWW, DD);
  gemm_wmma1<<<dim3(HH / 16, TT / 16), wave, 0, stream>>>(h_b, wbT, braw, TT, HH, DD);
  gemm_wmma4<<<dim3(LOWW / 64, TT / 16), wave, 0, stream>>>(h_b, wgaT, gaf, TT, LOWW, DD);
  cvt_bf16_kernel<<<(TT * LOWW + 255) / 256, 256, 0, stream>>>(gaf, ga_b, TT * LOWW);
  gemm_wmma4<<<dim3(DD / 64, TT / 16), wave, 0, stream>>>(ga_b, wgbT, gate, TT, DD, LOWW);

  // ---- 3. causal conv + silu, l2norm, gates ----
  conv_silu_kernel<<<(nTD + 255) / 256, 256, 0, stream>>>(qpre, conv_q, qc, DD);
  conv_silu_kernel<<<(nTD + 255) / 256, 256, 0, stream>>>(kpre, conv_k, kc, DD);
  conv_silu_kernel<<<(nTD + 255) / 256, 256, 0, stream>>>(vpre, conv_v, vc, DD);
  l2norm_kernel<<<TT * HH, 128, 0, stream>>>(qc, 0.08838834764831845f); // DK^-0.5
  l2norm_kernel<<<TT * HH, 128, 0, stream>>>(kc, 1.0f);
  gbeta_kernel<<<(TT * HH + 255) / 256, 256, 0, stream>>>(fa, wfb, dt_bias, a_log, braw, garr, barr);

  // ---- 4. gated delta-rule recurrent scan (one head per WGP, S in LDS) ----
  kda_scan_kernel<<<HH, 256, 0, stream>>>(qc, kc, vc, garr, barr, o_s);

  // ---- 5. gated RMSNorm + output projection + residual ----
  gated_out_kernel<<<TT * HH, 128, 0, stream>>>(o_s, o_norm_w, gate, ogt_b);
  gemm_wmma4<<<dim3(DD / 64, TT / 16), wave, 0, stream>>>(ogt_b, woT, attn, TT, DD, DD);
  add_kernel<<<(nTD + 255) / 256, 256, 0, stream>>>(x, attn, x2, nTD);

  // ---- 6. MLP: RMSNorm, fused SwiGLU dual-GEMM, down proj, residual ----
  rmsnorm_kernel<<<TT, 256, 0, stream>>>(x2, ln2_w, h2_b);
  gemm_swiglu_wmma<<<dim3(II / 32, TT / 16), wave, 0, stream>>>(h2_b, wgT, wuT, P_b, TT, II, DD);
  gemm_wmma4<<<dim3(DD / 64, TT / 16), wave, 0, stream>>>(P_b, wdT, mlp, TT, DD, II);
  add_kernel<<<(nTD + 255) / 256, 256, 0, stream>>>(x2, mlp, out, nTD);
}
